// RNN_6579889898094
// MI455X (gfx1250) — matrix operand order repaired and checked by reading, not compiled, not run on hardware
//
#include <hip/hip_runtime.h>
#include <hip/hip_bf16.h>
#include <math.h>

#define B_ 128
#define T_ 256
#define I_ 256
#define H_ 1024
#define O_ 128

typedef _Float16 v16h __attribute__((ext_vector_type(16)));
typedef _Float16 h8   __attribute__((ext_vector_type(8)));
typedef _Float16 h4   __attribute__((ext_vector_type(4)));
typedef float    v8f  __attribute__((ext_vector_type(8)));

#if __has_builtin(__builtin_amdgcn_tanhf)
#define TANHF(x) __builtin_amdgcn_tanhf(x)
#else
#define TANHF(x) tanhf(x)
#endif

__device__ __forceinline__ v8f wmma_f16(v16h a, v16h b, v8f c) {
  // D = A*B + C, f16 inputs, f32 accumulate.
  return __builtin_amdgcn_wmma_f32_16x16x32_f16(false, a, false, b, (short)0, c,
                                                false, false);
}

// gfx1250 async DMA global<->LDS (16B per lane), tracked by ASYNCcnt.
// LDS operand = LDS byte offset (low 32 bits of generic shared address).
__device__ __forceinline__ void async_copy_b128(void* lds, const void* g) {
  asm volatile("global_load_async_to_lds_b128 %0, %1, off"
               :
               : "v"((unsigned)(size_t)lds), "v"(g)
               : "memory");
}
__device__ __forceinline__ void async_store_b128(void* g, const void* lds) {
  asm volatile("global_store_async_from_lds_b128 %0, %1, off"
               :
               : "v"(g), "v"((unsigned)(size_t)lds)
               : "memory");
}
__device__ __forceinline__ void wait_async0() {
  asm volatile("s_wait_asynccnt 0x0" ::: "memory");
}

// B-matrix fragment (32x16 tile of W^T) straight from row-major W (N x K):
// lane holds column n = n0 + (lane&15), K block 16*(lane>>4)..+15
// -> 16 contiguous f16 -> one clause of b128 loads.
__device__ __forceinline__ v16h load_bfrag_g(const _Float16* __restrict__ W,
                                             int n0, int k0, int ldk, int lane) {
  int nn = n0 + (lane & 15);
  int kk = k0 + ((lane >> 4) << 4);
  const h8* p = (const h8*)(W + (size_t)nn * ldk + kk);
  union { v16h v; h8 h[2]; } u;
  u.h[0] = p[0];
  u.h[1] = p[1];
  return u.v;
}

// A-matrix fragment (16x32 f16) from LDS, row stride ldk (padded).
// lane: m = lane&15, half = lane>>4; VGPR0-7: K=16*half+0..15
// -> two 16B ds loads.
__device__ __forceinline__ v16h load_afrag_lds(const _Float16* base, int ldk,
                                               int k0, int lane) {
  int m = lane & 15;
  int half = lane >> 4;
  const _Float16* row = base + m * ldk + k0 + 16 * half;
  union { v16h v; h8 h[2]; } u;
  u.h[0] = *(const h8*)(row);
  u.h[1] = *(const h8*)(row + 8);
  return u.v;
}

__global__ void cvt_f32_to_f16_kernel(const float* __restrict__ s,
                                      _Float16* __restrict__ d, int n) {
  int i = blockIdx.x * blockDim.x + threadIdx.x;
  if (i < n) d[i] = (_Float16)s[i];
}

__global__ void bias_sum_kernel(const float* __restrict__ a,
                                const float* __restrict__ b,
                                float* __restrict__ o, int n) {
  int i = blockIdx.x * blockDim.x + threadIdx.x;
  if (i < n) o[i] = a[i] + b[i];
}

// GEMM: out[M,N] = A[M,K] @ W[N,K]^T + biasA(+biasB); f16 math, f32 accum.
// Block: 256 threads (8 waves), tile M=32, N=256. A chunk staged in LDS.
template <bool AF16, bool OUTF16>
__global__ __launch_bounds__(256) void gemm_wmma_kernel(
    const void* __restrict__ Ap, const _Float16* __restrict__ W,
    const float* __restrict__ biasA, const float* __restrict__ biasB,
    void* __restrict__ outp, int M, int N, int K) {
  __shared__ _Float16 a_lds[2][16][40];  // +8 pad: conflict-free b128 reads
  const int tid = threadIdx.x;
  const int lane = tid & 31;
  const int wave = tid >> 5;
  const int half = lane >> 4;
  const int mt = wave & 1;
  const int ng = wave >> 1;
  const int m0 = blockIdx.y * 32;
  const int n0 = blockIdx.x * 256 + ng * 64;
  const bool active = (n0 + 64) <= N;  // wave-uniform

  v8f c[4] = {};

  const int srow = tid >> 3;      // 0..31: A row within tile
  const int skq = (tid & 7) * 4;  // 0..28: K quad

  for (int k0 = 0; k0 < K; k0 += 32) {
    __syncthreads();
    if (AF16) {
      const _Float16* A16 = (const _Float16*)Ap;
      h4 v = *(const h4*)(A16 + (size_t)(m0 + srow) * K + k0 + skq);
      *(h4*)&a_lds[srow >> 4][srow & 15][skq] = v;
    } else {
      const float* A32 = (const float*)Ap;
      float4 v = *(const float4*)(A32 + (size_t)(m0 + srow) * K + k0 + skq);
      h4 hv = {(_Float16)v.x, (_Float16)v.y, (_Float16)v.z, (_Float16)v.w};
      *(h4*)&a_lds[srow >> 4][srow & 15][skq] = hv;
    }
    __syncthreads();
    if (active) {
      v16h a = load_afrag_lds(&a_lds[mt][0][0], 40, 0, lane);
#pragma unroll
      for (int f = 0; f < 4; ++f) {
        v16h b = load_bfrag_g(W, n0 + 16 * f, k0, K, lane);
        c[f] = wmma_f16(a, b, c[f]);
      }
    }
  }
  if (active) {
    const int nn = lane & 15;
#pragma unroll
    for (int f = 0; f < 4; ++f) {
      int n = n0 + 16 * f + nn;
      float bb = biasA ? biasA[n] : 0.0f;
      if (biasB) bb += biasB[n];
#pragma unroll
      for (int r = 0; r < 8; ++r) {
        int m = m0 + mt * 16 + r + 8 * half;
        if (OUTF16) {
          ((_Float16*)outp)[(size_t)m * N + n] = (_Float16)(c[f][r] + bb);
        } else {
          ((float*)outp)[(size_t)m * N + n] = c[f][r] + bb;
        }
      }
    }
  }
}

// Persistent recurrent scan. Each block owns 16 batch rows (rows are
// independent in the recurrence). h (16x1024 f16) lives in LDS; 32 waves,
// wave w computes output columns [32w, 32w+32).
// Pipeline per step (ASYNCcnt):
//   issue async load of xw tile t+1 -> WMMA loop -> s_wait_asynccnt 0 ->
//   barrier -> write new h to LDS -> barrier -> async store h_t to global.
__global__ __launch_bounds__(1024) void rnn_scan_kernel(
    const _Float16* __restrict__ xw, const _Float16* __restrict__ Whh,
    _Float16* __restrict__ h_all, _Float16* __restrict__ h_last) {
  constexpr int HS = H_ + 8;  // pad -> conflict-free A-frag ds_load_b128
  extern __shared__ char smem[];
  _Float16* hbuf = (_Float16*)smem;                          // 16*HS
  _Float16* xwt0 = (_Float16*)(smem + (size_t)16 * HS * 2);  // 2 tiles 16*H_

  const int tid = threadIdx.x;
  const int lane = tid & 31;
  const int wave = tid >> 5;
  const int half = lane >> 4;
  const int b0 = blockIdx.x * 16;
  const int n0 = wave * 32;

  // Async staging slice for this thread: row = tid>>6, 32B at col (tid&63)*16
  const int crow = tid >> 6;
  const int ccol = (tid & 63) * 16;

  for (int i = tid; i < 16 * HS; i += 1024) hbuf[i] = (_Float16)0.0f;

  {  // stage tile t=0
    const _Float16* src = xw + (((size_t)(b0 + crow)) * T_ + 0) * H_ + ccol;
    _Float16* dst = xwt0 + crow * H_ + ccol;
    async_copy_b128(dst, src);
    async_copy_b128(dst + 8, src + 8);
  }
  wait_async0();
  __syncthreads();

  for (int t = 0; t < T_; ++t) {
    if (t + 1 < T_) {  // overlap next tile's copy with this step's WMMAs
      const _Float16* src =
          xw + (((size_t)(b0 + crow)) * T_ + (t + 1)) * H_ + ccol;
      _Float16* dst = xwt0 + (size_t)((t + 1) & 1) * (16 * H_) + crow * H_ + ccol;
      async_copy_b128(dst, src);
      async_copy_b128(dst + 8, src + 8);
    }
    v8f c0 = {};
    v8f c1 = {};
#pragma unroll 4
    for (int kk = 0; kk < H_; kk += 32) {
      v16h a = load_afrag_lds(hbuf, HS, kk, lane);
      v16h bf0 = load_bfrag_g(Whh, n0, kk, H_, lane);
      v16h bf1 = load_bfrag_g(Whh, n0 + 16, kk, H_, lane);
      c0 = wmma_f16(a, bf0, c0);
      c1 = wmma_f16(a, bf1, c1);
    }
    float vals[16];
    const int nbase = n0 + (lane & 15);
    const _Float16* xt = xwt0 + (size_t)(t & 1) * (16 * H_);  // LDS-inferable
#pragma unroll
    for (int r = 0; r < 8; ++r) {
      int m = r + 8 * half;
      vals[r] = TANHF(c0[r] + (float)xt[m * H_ + nbase]);
      vals[8 + r] = TANHF(c1[r] + (float)xt[m * H_ + nbase + 16]);
    }
    // Retire this thread's outstanding asyncs (tile t+1 load, spill of t-1),
    // then barrier so EVERY thread's asyncs are done before hbuf is rewritten.
    wait_async0();
    __syncthreads();
#pragma unroll
    for (int r = 0; r < 8; ++r) {
      int m = r + 8 * half;
      hbuf[m * HS + nbase] = (_Float16)vals[r];
      hbuf[m * HS + nbase + 16] = (_Float16)vals[8 + r];
    }
    __syncthreads();  // new h visible (ds drained by barrier)
    if (h_all) {      // async DMA spill LDS -> global, overlaps next step
      _Float16* d = &h_all[(((size_t)(b0 + crow)) * T_ + t) * H_ + ccol];
      const _Float16* s = &hbuf[crow * HS + ccol];
      async_store_b128(d, s);
      async_store_b128(d + 8, s + 8);
    }
    if (h_last && t == T_ - 1) {
      _Float16* d = &h_last[(size_t)(b0 + crow) * H_ + ccol];
      const _Float16* s = &hbuf[crow * HS + ccol];
      async_store_b128(d, s);
      async_store_b128(d + 8, s + 8);
    }
  }
  wait_async0();  // drain final spill (S_ENDPGM would also wait-idle)
}

extern "C" void kernel_launch(void* const* d_in, const int* in_sizes, int n_in,
                              void* d_out, int out_size, void* d_ws,
                              size_t ws_size, hipStream_t stream) {
  (void)in_sizes; (void)n_in; (void)out_size; (void)ws_size;
  const float* X = (const float*)d_in[0];
  const float* W_ih0 = (const float*)d_in[1];
  const float* W_hh0 = (const float*)d_in[2];
  const float* b_ih0 = (const float*)d_in[3];
  const float* b_hh0 = (const float*)d_in[4];
  const float* W_ih1 = (const float*)d_in[5];
  const float* W_hh1 = (const float*)d_in[6];
  const float* b_ih1 = (const float*)d_in[7];
  const float* b_hh1 = (const float*)d_in[8];
  const float* W_fc = (const float*)d_in[9];
  const float* b_fc = (const float*)d_in[10];
  float* out = (float*)d_out;

  char* ws = (char*)d_ws;
  size_t off = 0;
  auto alloc = [&](size_t bytes) {
    char* p = ws + off;
    off = (off + bytes + 255) & ~(size_t)255;
    return p;
  };
  _Float16* xwbuf = (_Float16*)alloc((size_t)B_ * T_ * H_ * 2);  // 67 MB
  _Float16* h1buf = (_Float16*)alloc((size_t)B_ * T_ * H_ * 2);  // 67 MB
  _Float16* wih0h = (_Float16*)alloc((size_t)H_ * I_ * 2);
  _Float16* whh0h = (_Float16*)alloc((size_t)H_ * H_ * 2);
  _Float16* wih1h = (_Float16*)alloc((size_t)H_ * H_ * 2);
  _Float16* whh1h = (_Float16*)alloc((size_t)H_ * H_ * 2);
  _Float16* wfch = (_Float16*)alloc((size_t)O_ * H_ * 2);
  _Float16* hlast = (_Float16*)alloc((size_t)B_ * H_ * 2);
  float* bias0 = (float*)alloc(H_ * 4);
  float* bias1 = (float*)alloc(H_ * 4);

  // Weight conversion f32 -> f16 (one-time, tiny)
  cvt_f32_to_f16_kernel<<<(H_ * I_ + 255) / 256, 256, 0, stream>>>(W_ih0, wih0h, H_ * I_);
  cvt_f32_to_f16_kernel<<<(H_ * H_ + 255) / 256, 256, 0, stream>>>(W_hh0, whh0h, H_ * H_);
  cvt_f32_to_f16_kernel<<<(H_ * H_ + 255) / 256, 256, 0, stream>>>(W_ih1, wih1h, H_ * H_);
  cvt_f32_to_f16_kernel<<<(H_ * H_ + 255) / 256, 256, 0, stream>>>(W_hh1, whh1h, H_ * H_);
  cvt_f32_to_f16_kernel<<<(O_ * H_ + 255) / 256, 256, 0, stream>>>(W_fc, wfch, O_ * H_);
  bias_sum_kernel<<<(H_ + 255) / 256, 256, 0, stream>>>(b_ih0, b_hh0, bias0, H_);
  bias_sum_kernel<<<(H_ + 255) / 256, 256, 0, stream>>>(b_ih1, b_hh1, bias1, H_);

  constexpr size_t SMEM_SCAN = ((size_t)16 * (H_ + 8) + (size_t)32 * H_) * 2;  // 98.5 KB

  // Layer 0 input projection: xw0 = X @ W_ih0^T + (b_ih0+b_hh0)  (f16 out)
  gemm_wmma_kernel<false, true><<<dim3((H_ + 255) / 256, (B_ * T_) / 32), 256, 0, stream>>>(
      X, wih0h, bias0, nullptr, xwbuf, B_ * T_, H_, I_);
  // Layer 0 recurrence -> h1 (all timesteps, f16)
  rnn_scan_kernel<<<B_ / 16, 1024, SMEM_SCAN, stream>>>(xwbuf, whh0h, h1buf, nullptr);
  // Layer 1 input projection: xw1 = h1 @ W_ih1^T + (b_ih1+b_hh1)  (f16 out)
  gemm_wmma_kernel<true, true><<<dim3((H_ + 255) / 256, (B_ * T_) / 32), 256, 0, stream>>>(
      h1buf, wih1h, bias1, nullptr, xwbuf, B_ * T_, H_, H_);
  // Layer 1 recurrence -> only final hidden state
  rnn_scan_kernel<<<B_ / 16, 1024, SMEM_SCAN, stream>>>(xwbuf, whh1h, nullptr, hlast);
  // FC: out = h_last @ W_fc^T + b_fc  (f32 out -> d_out)
  gemm_wmma_kernel<true, false><<<dim3((O_ + 255) / 256, B_ / 32), 256, 0, stream>>>(
      hlast, wfch, b_fc, nullptr, out, B_, O_, H_);
}
